// VQEmbedding_16862041604579
// MI455X (gfx1250) — matrix operand order repaired and checked by reading, not compiled, not run on hardware
//
#include <hip/hip_runtime.h>
#include <hip/hip_bf16.h>

// ---- problem constants (fixed by reference setup_inputs) ----
#define KC   512          // number of codes
#define DD   64           // embedding dim
#define NPTS 65536        // 16*64*64 points
#define HW   4096         // 64*64
#define DECAYF 0.99f
#define ONEMD  0.01f
#define EPSF   1e-5f

typedef __attribute__((ext_vector_type(16))) __bf16 v16bf;
typedef __attribute__((ext_vector_type(8)))  float  v8f;

// --------------------------------------------------------------------------
// Kernel 1: prep — bf16 codebook copy, ||e||^2, zero counts/emb_sum
// grid: KC blocks x 32 threads
__global__ __launch_bounds__(32)
void vq_prep(const float* __restrict__ emb_w,
             __bf16* __restrict__ emb_bf,
             float* __restrict__ e2,
             float* __restrict__ counts,
             float* __restrict__ emb_sum) {
    const int k = blockIdx.x;
    const int lane = threadIdx.x;
    float s = 0.0f;
#pragma unroll
    for (int c = 0; c < 2; ++c) {
        const int d = c * 32 + lane;
        const float v = emb_w[k * DD + d];
        emb_bf[k * DD + d] = (__bf16)v;
        emb_sum[k * DD + d] = 0.0f;
        s += v * v;
    }
#pragma unroll
    for (int m = 1; m < 32; m <<= 1) s += __shfl_xor(s, m);
    if (lane == 0) { e2[k] = s; counts[k] = 0.0f; }
}

// --------------------------------------------------------------------------
// Kernel 2: assignment via bf16 WMMA distance GEMM + argmin.
// Each wave owns TWO 16-point A tiles (M=32) so every B chunk staged from
// LDS feeds 2 WMMAs (4 WMMAs per 16-code iteration).
// grid: NPTS/32/8 = 256 blocks x 256 threads
// dynamic LDS: KC*DD bf16 (64KB) + KC floats (2KB)
__global__ __launch_bounds__(256)
void vq_assign(const float* __restrict__ z,
               const __bf16* __restrict__ emb_bf,
               const float* __restrict__ e2g,
               int* __restrict__ idx_out,
               float* __restrict__ counts) {
    extern __shared__ char smem[];
    __bf16* embLds = (__bf16*)smem;
    float*  e2s    = (float*)(smem + (size_t)KC * DD * sizeof(__bf16));

    // cooperative stage of codebook + norms into LDS
    {
        const uint32_t* s32 = (const uint32_t*)emb_bf;
        uint32_t* d32 = (uint32_t*)embLds;
        for (int i = threadIdx.x; i < KC * DD / 2; i += 256) d32[i] = s32[i];
        for (int i = threadIdx.x; i < KC; i += 256) e2s[i] = e2g[i];
    }
    __syncthreads();

    const int wave = threadIdx.x >> 5;
    const int lane = threadIdx.x & 31;
    const int half = lane >> 4;       // which 16-lane half
    const int mrow = lane & 15;       // row (for A) / col (for B,C)
    const int tile0 = (blockIdx.x * 8 + wave) * 2;   // first of two 16-pt tiles

    // ---- load two A tiles (2 x 16 pts x 64 dims) per 16-bit A VGPR layout --
    // point p, dim d lives at z[(b*64+d)*4096 + hw]
    v16bf a0[2], a1[2];
    float x2row0[8], x2row1[8];
#pragma unroll
    for (int tt = 0; tt < 2; ++tt) {
        const int p  = (tile0 + tt) * 16 + mrow;
        const int b  = p >> 12;
        const int hw = p & (HW - 1);
        const float* zb = z + (size_t)b * DD * HW + hw;
        float x2p = 0.0f;
#pragma unroll
        for (int c = 0; c < 2; ++c) {
#pragma unroll
            for (int j = 0; j < 16; ++j) {
                const int d = c * 32 + half * 8 + (j < 8 ? j : j + 8);
                const float v = zb[(size_t)d * HW];
                x2p += v * v;
                if (tt == 0) a0[c][j] = (__bf16)v;
                else         a1[c][j] = (__bf16)v;
            }
        }
        // lane l and l^16 hold complementary dim-halves of row m = l%16
        const float x2own = x2p + __shfl_xor(x2p, 16);
#pragma unroll
        for (int r = 0; r < 8; ++r) {
            const float xr = __shfl(x2own, half * 8 + r);
            if (tt == 0) x2row0[r] = xr; else x2row1[r] = xr;
        }
    }

    float best0[8], best1[8];
    int   bcode0[8], bcode1[8];
#pragma unroll
    for (int r = 0; r < 8; ++r) {
        best0[r] = 3.0e38f; bcode0[r] = 0;
        best1[r] = 3.0e38f; bcode1[r] = 0;
    }

    // ---- sweep all 512 codes in 16-code tiles ----
    for (int t = 0; t < KC / 16; ++t) {
        const int n = t * 16 + mrow;           // this lane's code column
        const float e2n = e2s[n];
        // B (32x16): lanes 0-15 hold k=0..7,16..23, lanes 16-31 hold k=8..15,24..31
        v16bf bm[2];
#pragma unroll
        for (int c = 0; c < 2; ++c) {
            const __bf16* bs = embLds + n * DD + c * 32 + half * 8;
#pragma unroll
            for (int j = 0; j < 16; ++j) bm[c][j] = bs[j < 8 ? j : j + 8];
        }
        v8f acc0 = {}, acc1 = {};
#pragma unroll
        for (int c = 0; c < 2; ++c) {
            acc0 = __builtin_amdgcn_wmma_f32_16x16x32_bf16(
                false, a0[c], false, bm[c], (short)0, acc0, false, false);
            acc1 = __builtin_amdgcn_wmma_f32_16x16x32_bf16(
                false, a1[c], false, bm[c], (short)0, acc1, false, false);
        }
        // C layout: element r = G[row r+8*half][col mrow]
#pragma unroll
        for (int r = 0; r < 8; ++r) {
            const float d0 = x2row0[r] + e2n - 2.0f * acc0[r];
            if (d0 < best0[r]) { best0[r] = d0; bcode0[r] = n; }
            const float d1 = x2row1[r] + e2n - 2.0f * acc1[r];
            if (d1 < best1[r]) { best1[r] = d1; bcode1[r] = n; }
        }
    }

    // ---- butterfly min-reduce over the 16 columns in each half ----
#pragma unroll
    for (int s = 0; s < 4; ++s) {
        const int mask = 1 << s;
#pragma unroll
        for (int r = 0; r < 8; ++r) {
            {
                const float ob = __shfl_xor(best0[r], mask);
                const int   oc = __shfl_xor(bcode0[r], mask);
                if (ob < best0[r] || (ob == best0[r] && oc < bcode0[r])) {
                    best0[r] = ob; bcode0[r] = oc;
                }
            }
            {
                const float ob = __shfl_xor(best1[r], mask);
                const int   oc = __shfl_xor(bcode1[r], mask);
                if (ob < best1[r] || (ob == best1[r] && oc < bcode1[r])) {
                    best1[r] = ob; bcode1[r] = oc;
                }
            }
        }
    }

    if (mrow == 0) {
#pragma unroll
        for (int r = 0; r < 8; ++r) {
            const int m = half * 8 + r;
            idx_out[tile0 * 16 + m] = bcode0[r];
            atomicAdd(&counts[bcode0[r]], 1.0f);
            idx_out[(tile0 + 1) * 16 + m] = bcode1[r];
            atomicAdd(&counts[bcode1[r]], 1.0f);
        }
    }
}

// --------------------------------------------------------------------------
// Kernel 3: emb_sum[k] += flat[p]  (segment sum via f32 atomics, L2-resident)
// grid: NPTS/256 blocks x 256 threads
__global__ __launch_bounds__(256)
void vq_scatter(const float* __restrict__ z,
                const int* __restrict__ idx,
                float* __restrict__ emb_sum) {
    const int p  = blockIdx.x * 256 + threadIdx.x;
    const int k  = idx[p];
    const int b  = p >> 12;
    const int hw = p & (HW - 1);
    const float* zb = z + (size_t)b * DD * HW + hw;
#pragma unroll 4
    for (int d = 0; d < DD; ++d)
        atomicAdd(&emb_sum[k * DD + d], zb[(size_t)d * HW]);
}

// --------------------------------------------------------------------------
// Kernel 4: EMA update + Laplace smoothing -> new codebook; zero loss slot
// grid: 1 block x 512 threads
__global__ __launch_bounds__(512)
void vq_update(const float* __restrict__ cluster_size,
               const float* __restrict__ counts,
               const float* __restrict__ ema_w,
               const float* __restrict__ emb_sum,
               float* __restrict__ new_emb,
               float* __restrict__ loss_out) {
    __shared__ float red[KC];
    const int k = threadIdx.x;
    const float cs = cluster_size[k] * DECAYF + ONEMD * counts[k];
    red[k] = cs;
    __syncthreads();
    for (int s = 256; s > 0; s >>= 1) {
        if (k < s) red[k] += red[k + s];
        __syncthreads();
    }
    const float n = red[0];
    const float css = (cs + EPSF) / (n + (float)KC * EPSF) * n;
    const float inv = 1.0f / css;
#pragma unroll 4
    for (int d = 0; d < DD; ++d) {
        const int i = k * DD + d;
        new_emb[i] = (ema_w[i] * DECAYF + ONEMD * emb_sum[i]) * inv;
    }
    if (k == 0) *loss_out = 0.0f;
}

// --------------------------------------------------------------------------
// Kernel 5: gather updated codes into [B,D,H,W] output + MSE loss
// grid: NPTS/256 blocks x 256 threads
__global__ __launch_bounds__(256)
void vq_gather(const float* __restrict__ z,
               const int* __restrict__ idx,
               const float* __restrict__ new_emb,
               float* __restrict__ out,
               float* __restrict__ loss_out) {
    const int p  = blockIdx.x * 256 + threadIdx.x;
    const int k  = idx[p];
    const int b  = p >> 12;
    const int hw = p & (HW - 1);
    const float* zb = z + (size_t)b * DD * HW + hw;
    float* ob = out + (size_t)b * DD * HW + hw;
    float acc = 0.0f;
#pragma unroll 4
    for (int d = 0; d < DD; ++d) {
        const float q = new_emb[k * DD + d];
        const float x = zb[(size_t)d * HW];
        const float df = q - x;
        acc += df * df;
        ob[(size_t)d * HW] = q;   // straight-through value == quantized value
    }
#pragma unroll
    for (int s = 1; s < 32; s <<= 1) acc += __shfl_xor(acc, s);
    if ((threadIdx.x & 31) == 0)
        atomicAdd(loss_out, acc * (1.0f / (float)(NPTS * DD)));
}

// --------------------------------------------------------------------------
extern "C" void kernel_launch(void* const* d_in, const int* in_sizes, int n_in,
                              void* d_out, int out_size, void* d_ws, size_t ws_size,
                              hipStream_t stream) {
    const float* z            = (const float*)d_in[0];   // [16,64,64,64]
    const float* emb_w        = (const float*)d_in[1];   // [512,64]
    const float* ema_w        = (const float*)d_in[2];   // [512,64]
    const float* cluster_size = (const float*)d_in[3];   // [512]
    float* out = (float*)d_out;                          // [16,64,64,64] + 1 loss

    // workspace layout (bytes)
    char* ws = (char*)d_ws;
    int*    w_idx     = (int*)(ws);                            // 65536 * 4
    float*  w_counts  = (float*)(ws + 262144);                 // 512 * 4
    float*  w_e2      = (float*)(ws + 264192);                 // 512 * 4
    float*  w_embsum  = (float*)(ws + 266240);                 // 512*64*4
    float*  w_newemb  = (float*)(ws + 397312);                 // 512*64*4
    __bf16* w_embbf   = (__bf16*)(ws + 528384);                // 512*64*2

    float* loss_slot = out + (size_t)NPTS * DD;                // last element

    vq_prep<<<KC, 32, 0, stream>>>(emb_w, w_embbf, w_e2, w_counts, w_embsum);

    const size_t lds_bytes = (size_t)KC * DD * sizeof(__bf16) + KC * sizeof(float);
    vq_assign<<<NPTS / 32 / 8, 256, lds_bytes, stream>>>(
        z, w_embbf, w_e2, w_idx, w_counts);

    vq_scatter<<<NPTS / 256, 256, 0, stream>>>(z, w_idx, w_embsum);

    vq_update<<<1, KC, 0, stream>>>(cluster_size, w_counts, ema_w, w_embsum,
                                    w_newemb, loss_slot);

    vq_gather<<<NPTS / 256, 256, 0, stream>>>(z, w_idx, w_newemb, out, loss_slot);
}
